// vaenc_58445914964279
// MI455X (gfx1250) — matrix operand order repaired and checked by reading, not compiled, not run on hardware
//
#include <hip/hip_runtime.h>

typedef unsigned short u16;
typedef __attribute__((ext_vector_type(16))) __bf16 v16bf;
typedef __attribute__((ext_vector_type(8)))  float   v8f;

// ---------------- helpers ----------------

__device__ __forceinline__ u16 f2bf(float f) {
  union { float f; unsigned u; } v; v.f = f;
  unsigned u = v.u;
  unsigned r = u + 0x7fffu + ((u >> 16) & 1u);   // RNE
  return (u16)(r >> 16);
}

__device__ __forceinline__ float logsoftplus(float v) {
  float sp = (v > 15.f) ? v : log1pf(expf(v));
  return logf(sp);
}

// async global->LDS copy of 16 bytes (per lane). ldsoff = wave-relative LDS
// byte offset (= low 32 bits of the flat LDS address), g = global src.
__device__ __forceinline__ void async_copy_b128(unsigned ldsoff, const void* g) {
  asm volatile("global_load_async_to_lds_b128 %0, %1, off"
               :: "v"(ldsoff), "v"(g) : "memory");
}
__device__ __forceinline__ void wait_asynccnt0() {
  asm volatile("s_wait_asynccnt 0" ::: "memory");
}

enum { EP_F32 = 0, EP_BIAS_F32 = 1, EP_LOGSP_F32 = 2, EP_RELU_BF16 = 3 };

// ---------------- WMMA GEMM ----------------
// C[M,N] = epilogue( A[M,K](bf16,row-major) * B^T  + bias )
// Bt is stored [N,K] row-major (the transpose of the mathematical B), so both
// A and B fragments are contiguous 16B reads from LDS in a contiguous k
// order (k = 16*half + s), the same in A and B. Tiles are staged with
// GLOBAL_LOAD_ASYNC_TO_LDS_B128 into double-buffered LDS so the DMA for
// K-tile k+1 overlaps the WMMA stream on K-tile k.
template<int BM, int BN, int WM, int WN, int EP>
__global__ __launch_bounds__(WM*WN*32)
void gemm_bf16_wmma(const u16* __restrict__ A, const u16* __restrict__ Bt,
                    const float* __restrict__ bias, void* __restrict__ outp,
                    int Mdim, int Ndim, int Kdim)
{
  constexpr int KT  = 64;          // K-depth per LDS tile (2 WMMA steps)
  constexpr int LDT = KT + 8;      // padded row stride (72 elems = 144B, 16B-aligned)
  constexpr int NT  = WM*WN*32;    // threads per block
  constexpr int WTM = BM / WM;     // wave tile M
  constexpr int WTN = BN / WN;     // wave tile N
  constexpr int MF  = WTM / 16;
  constexpr int NF  = WTN / 16;

  __shared__ __align__(16) u16 sA[2][BM * LDT];
  __shared__ __align__(16) u16 sB[2][BN * LDT];

  const int tid  = threadIdx.x;
  const int wave = tid >> 5;
  const int lane = tid & 31;
  const int wm   = wave / WN;
  const int wn   = wave % WN;
  const int lh   = lane >> 4;   // lane half (0/1)
  const int ln   = lane & 15;

  const int m0 = blockIdx.x * BM;
  const int n0 = blockIdx.y * BN;

  v8f acc[MF][NF];
  #pragma unroll
  for (int i = 0; i < MF; ++i)
    #pragma unroll
    for (int j = 0; j < NF; ++j) {
      v8f z = {0.f,0.f,0.f,0.f,0.f,0.f,0.f,0.f};
      acc[i][j] = z;
    }

  // issue async DMA for one K-tile into buffer `buf` (no wait)
  auto issue_tile = [&](int k0, int buf) {
    #pragma unroll
    for (int c = tid; c < BM*KT/8; c += NT) {
      int r = c >> 3, cc = (c & 7) * 8;
      async_copy_b128((unsigned)(size_t)&sA[buf][r*LDT + cc],
                      &A[(size_t)(m0 + r)*Kdim + (size_t)(k0 + cc)]);
    }
    #pragma unroll
    for (int c = tid; c < BN*KT/8; c += NT) {
      int r = c >> 3, cc = (c & 7) * 8;
      async_copy_b128((unsigned)(size_t)&sB[buf][r*LDT + cc],
                      &Bt[(size_t)(n0 + r)*Kdim + (size_t)(k0 + cc)]);
    }
  };

  issue_tile(0, 0);
  wait_asynccnt0();
  __syncthreads();

  int buf = 0;
  for (int k0 = 0; k0 < Kdim; k0 += KT, buf ^= 1) {
    if (k0 + KT < Kdim) issue_tile(k0 + KT, buf ^ 1);   // overlap DMA w/ compute

    const u16* cA = sA[buf];
    const u16* cB = sB[buf];
    #pragma unroll
    for (int kk = 0; kk < KT; kk += 32) {
      union Frag { v16bf v; uint4 q[2]; };
      // B fragments: lane<16 -> N=ln, K=kk..kk+15 ; lane>=16 -> K=kk+16..kk+31
      Frag bf[NF];
      #pragma unroll
      for (int j = 0; j < NF; ++j) {
        const u16* p = &cB[(wn*WTN + j*16 + ln)*LDT + kk + lh*16];
        bf[j].q[0] = *(const uint4*)p;
        bf[j].q[1] = *(const uint4*)(p + 8);
      }
      #pragma unroll
      for (int i = 0; i < MF; ++i) {
        // A fragment: lane half selects K 0-15 vs 16-31
        Frag af;
        const u16* p = &cA[(wm*WTM + i*16 + ln)*LDT + kk + lh*16];
        af.q[0] = *(const uint4*)p;          // K = kk + lh*16 + 0..7
        af.q[1] = *(const uint4*)(p + 8);    // K = kk + lh*16 + 8..15
        #pragma unroll
        for (int j = 0; j < NF; ++j) {
          acc[i][j] = __builtin_amdgcn_wmma_f32_16x16x32_bf16(
              false, af.v, false, bf[j].v, (short)0, acc[i][j], false, false);
        }
      }
    }

    // next tile's DMA must be complete AND all waves done reading `buf`
    wait_asynccnt0();
    __syncthreads();
  }

  // epilogue: D layout -> row = base + lh*8 + v, col = ln
  #pragma unroll
  for (int i = 0; i < MF; ++i) {
    #pragma unroll
    for (int j = 0; j < NF; ++j) {
      const int gn = n0 + wn*WTN + j*16 + ln;
      const float bv = (EP == EP_F32) ? 0.f : bias[gn];
      const int mb = m0 + wm*WTM + i*16 + lh*8;
      #pragma unroll
      for (int v = 0; v < 8; ++v) {
        float x = acc[i][j][v] + bv;
        size_t idx = (size_t)(mb + v)*Ndim + gn;
        if (EP == EP_RELU_BF16) {
          ((u16*)outp)[idx] = f2bf(x > 0.f ? x : 0.f);
        } else if (EP == EP_LOGSP_F32) {
          ((float*)outp)[idx] = logsoftplus(x);
        } else {
          ((float*)outp)[idx] = x;
        }
      }
    }
  }
}

// ---------------- elementwise kernels ----------------

__global__ void k_convert_bf16(const float* __restrict__ in, u16* __restrict__ out, long n) {
  long i = (long)blockIdx.x*blockDim.x + threadIdx.x;
  if (i < n) out[i] = f2bf(in[i]);
}

// src [K,N] f32 -> dst [N,K] bf16
__global__ void k_transpose_bf16(const float* __restrict__ in, u16* __restrict__ out,
                                 int K, int N) {
  long i = (long)blockIdx.x*blockDim.x + threadIdx.x;
  if (i < (long)K*N) {
    int k = (int)(i / N), n = (int)(i % N);
    out[(size_t)n*K + k] = f2bf(in[i]);
  }
}

__global__ void k_bias_sum(const float* __restrict__ a, const float* __restrict__ b,
                           float* __restrict__ o, int n) {
  int i = blockIdx.x*blockDim.x + threadIdx.x;
  if (i < n) o[i] = a[i] + b[i];
}

__global__ void k_init_state(u16* __restrict__ h, float* __restrict__ c, int n) {
  int i = blockIdx.x*blockDim.x + threadIdx.x;
  if (i < n) { h[i] = 0; c[i] = 0.f; }
}

// z = qmu + eps * exp(0.5 * qlogvar)
__global__ void k_zrepar(const float* __restrict__ qmu, const float* __restrict__ qlv,
                         const float* __restrict__ eps, float* __restrict__ z, long n) {
  long i = (long)blockIdx.x*blockDim.x + threadIdx.x;
  if (i < n) z[i] = qmu[i] + eps[i] * expf(0.5f * qlv[i]);
}

// one block per batch row, reduce over L
__global__ void k_kl(const float* __restrict__ qmu, const float* __restrict__ qlv,
                     const float* __restrict__ pmu, const float* __restrict__ plv,
                     float* __restrict__ kl, int L) {
  __shared__ float red[256];
  int b = blockIdx.x;
  float s = 0.f;
  for (int l = threadIdx.x; l < L; l += blockDim.x) {
    size_t i = (size_t)b*L + l;
    float ql = qlv[i], pl = plv[i], dm = qmu[i] - pmu[i];
    s += 0.5f * (pl - ql + (expf(ql) + dm*dm) * expf(-pl) - 1.f);
  }
  red[threadIdx.x] = s; __syncthreads();
  for (int o = 128; o > 0; o >>= 1) {
    if ((int)threadIdx.x < o) red[threadIdx.x] += red[threadIdx.x + o];
    __syncthreads();
  }
  if (threadIdx.x == 0) kl[b] = red[0];
}

// gates = gx + gh (biases already folded into gx); split order i,f,g,o
__global__ void k_lstm(const float* __restrict__ gx, const float* __restrict__ gh,
                       float* __restrict__ c, u16* __restrict__ hbf, int H, int n) {
  int idx = blockIdx.x*blockDim.x + threadIdx.x;
  if (idx >= n) return;
  int b = idx / H, j = idx % H;
  size_t base = (size_t)b * 4 * H;
  float ig = gx[base + j]       + gh[base + j];
  float fg = gx[base + H + j]   + gh[base + H + j];
  float gg = gx[base + 2*H + j] + gh[base + 2*H + j];
  float og = gx[base + 3*H + j] + gh[base + 3*H + j];
  float si = 1.f / (1.f + expf(-ig));
  float sf = 1.f / (1.f + expf(-fg));
  float so = 1.f / (1.f + expf(-og));
  float cn = sf * c[idx] + si * tanhf(gg);
  float hn = so * tanhf(cn);
  c[idx] = cn;
  hbf[idx] = f2bf(hn);
}

// ---------------- host launcher ----------------

extern "C" void kernel_launch(void* const* d_in, const int* in_sizes, int n_in,
                              void* d_out, int out_size, void* d_ws, size_t ws_size,
                              hipStream_t stream) {
  (void)in_sizes; (void)n_in; (void)out_size; (void)ws_size;

  constexpr int T = 128, B = 256, D = 1024, H = 1024, L = 512, G = 4096;
  constexpr int M = T * B;                   // 32768 hoisted rows

  const float* sen   = (const float*)d_in[0];
  const float* eps   = (const float*)d_in[1];
  const float* Wq    = (const float*)d_in[2];
  const float* bq    = (const float*)d_in[3];
  const float* Wqmu  = (const float*)d_in[4];
  const float* bqmu  = (const float*)d_in[5];
  const float* Wqvar = (const float*)d_in[6];
  const float* bqvar = (const float*)d_in[7];
  const float* Wp    = (const float*)d_in[8];
  const float* bp    = (const float*)d_in[9];
  const float* Wpmu  = (const float*)d_in[10];
  const float* bpmu  = (const float*)d_in[11];
  const float* Wpvar = (const float*)d_in[12];
  const float* bpvar = (const float*)d_in[13];
  const float* Wx    = (const float*)d_in[14];
  const float* bx    = (const float*)d_in[15];
  const float* W_ih  = (const float*)d_in[16];
  const float* b_ih  = (const float*)d_in[17];
  const float* W_hh  = (const float*)d_in[18];
  const float* b_hh  = (const float*)d_in[19];

  float* zout  = (float*)d_out;                       // [T,B,L]
  float* klout = zout + (size_t)M * L;                // [T,B]

  // workspace bump allocator (256B aligned)
  char* p = (char*)d_ws;
  auto alloc = [&](size_t bytes) -> void* {
    void* r = (void*)p; p += (bytes + 255) & ~(size_t)255; return r;
  };
  u16*   Sbf    = (u16*)  alloc((size_t)M*D*2);
  u16*   Qbf    = (u16*)  alloc((size_t)M*D*2);
  u16*   Xbf    = (u16*)  alloc((size_t)M*H*2);
  float* QMU    = (float*)alloc((size_t)M*L*4);
  float* QLV    = (float*)alloc((size_t)M*L*4);
  float* GX     = (float*)alloc((size_t)M*G*4);       // x@W_ih^T + (b_ih+b_hh)
  u16*   WqT    = (u16*)  alloc((size_t)D*D*2);
  u16*   WqmuT  = (u16*)  alloc((size_t)D*L*2);
  u16*   WqvarT = (u16*)  alloc((size_t)D*L*2);
  u16*   WpT    = (u16*)  alloc((size_t)H*D*2);
  u16*   WpmuT  = (u16*)  alloc((size_t)D*L*2);
  u16*   WpvarT = (u16*)  alloc((size_t)D*L*2);
  u16*   WxT    = (u16*)  alloc((size_t)D*H*2);
  u16*   Wih    = (u16*)  alloc((size_t)G*H*2);       // already [N,K]
  u16*   Whh    = (u16*)  alloc((size_t)G*H*2);
  float* bsum   = (float*)alloc((size_t)G*4);
  u16*   Pbf    = (u16*)  alloc((size_t)B*D*2);
  float* PMU    = (float*)alloc((size_t)B*L*4);
  float* PLV    = (float*)alloc((size_t)B*L*4);
  float* GH     = (float*)alloc((size_t)B*G*4);
  u16*   Hbf    = (u16*)  alloc((size_t)B*H*2);
  float* Cf     = (float*)alloc((size_t)B*H*4);

  const int THR = 256;
  auto blks = [](long n, int t) { return (unsigned)((n + t - 1) / t); };

  // ---- phase 0: conversions / transposes (one-time per call) ----
  k_convert_bf16 <<<blks((long)M*D, THR), THR, 0, stream>>>(sen, Sbf, (long)M*D);
  k_transpose_bf16<<<blks((long)D*D, THR), THR, 0, stream>>>(Wq,    WqT,    D, D);
  k_transpose_bf16<<<blks((long)D*L, THR), THR, 0, stream>>>(Wqmu,  WqmuT,  D, L);
  k_transpose_bf16<<<blks((long)D*L, THR), THR, 0, stream>>>(Wqvar, WqvarT, D, L);
  k_transpose_bf16<<<blks((long)H*D, THR), THR, 0, stream>>>(Wp,    WpT,    H, D);
  k_transpose_bf16<<<blks((long)D*L, THR), THR, 0, stream>>>(Wpmu,  WpmuT,  D, L);
  k_transpose_bf16<<<blks((long)D*L, THR), THR, 0, stream>>>(Wpvar, WpvarT, D, L);
  k_transpose_bf16<<<blks((long)D*H, THR), THR, 0, stream>>>(Wx,    WxT,    D, H);
  k_convert_bf16 <<<blks((long)G*H, THR), THR, 0, stream>>>(W_ih, Wih, (long)G*H);
  k_convert_bf16 <<<blks((long)G*H, THR), THR, 0, stream>>>(W_hh, Whh, (long)G*H);
  k_bias_sum     <<<blks(G, THR), THR, 0, stream>>>(b_ih, b_hh, bsum, G);
  k_init_state   <<<blks((long)B*H, THR), THR, 0, stream>>>(Hbf, Cf, B*H);

  // ---- phase 1: hoisted q-branch over M = T*B rows (big WMMA GEMMs) ----
  gemm_bf16_wmma<128,128,2,4,EP_RELU_BF16><<<dim3(M/128, D/128), 256, 0, stream>>>(
      Sbf, WqT, bq, Qbf, M, D, D);
  gemm_bf16_wmma<128,128,2,4,EP_BIAS_F32><<<dim3(M/128, L/128), 256, 0, stream>>>(
      Qbf, WqmuT, bqmu, QMU, M, L, D);
  gemm_bf16_wmma<128,128,2,4,EP_LOGSP_F32><<<dim3(M/128, L/128), 256, 0, stream>>>(
      Qbf, WqvarT, bqvar, QLV, M, L, D);
  k_zrepar<<<blks((long)M*L, THR), THR, 0, stream>>>(QMU, QLV, eps, zout, (long)M*L);
  gemm_bf16_wmma<128,128,2,4,EP_RELU_BF16><<<dim3(M/128, H/128), 256, 0, stream>>>(
      Qbf, WxT, bx, Xbf, M, H, D);
  gemm_bf16_wmma<128,128,2,4,EP_BIAS_F32><<<dim3(M/128, G/128), 256, 0, stream>>>(
      Xbf, Wih, bsum, GX, M, G, H);

  // ---- phase 2: sequential recurrence (weights L2-resident: ~12MB bf16) ----
  for (int t = 0; t < T; ++t) {
    const float* qmu_t = QMU + (size_t)t*B*L;
    const float* qlv_t = QLV + (size_t)t*B*L;
    const float* gx_t  = GX  + (size_t)t*B*G;
    float*       kl_t  = klout + (size_t)t*B;

    gemm_bf16_wmma<64,64,2,2,EP_RELU_BF16><<<dim3(B/64, D/64), 128, 0, stream>>>(
        Hbf, WpT, bp, Pbf, B, D, H);
    gemm_bf16_wmma<64,64,2,2,EP_BIAS_F32><<<dim3(B/64, L/64), 128, 0, stream>>>(
        Pbf, WpmuT, bpmu, PMU, B, L, D);
    gemm_bf16_wmma<64,64,2,2,EP_LOGSP_F32><<<dim3(B/64, L/64), 128, 0, stream>>>(
        Pbf, WpvarT, bpvar, PLV, B, L, D);
    gemm_bf16_wmma<64,64,2,2,EP_F32><<<dim3(B/64, G/64), 128, 0, stream>>>(
        Hbf, Whh, nullptr, GH, B, G, H);

    k_kl  <<<B, 256, 0, stream>>>(qmu_t, qlv_t, PMU, PLV, kl_t, L);
    k_lstm<<<blks((long)B*H, THR), THR, 0, stream>>>(gx_t, GH, Cf, Hbf, H, B*H);
  }
}
